// Attention_79645873537262
// MI455X (gfx1250) — matrix operand order repaired and checked by reading, not compiled, not run on hardware
//
#include <hip/hip_runtime.h>

// ---------------------------------------------------------------------------
// Attention with pair bias, MI455X (gfx1250, wave32, WMMA, TDM, async-LDS).
// B=1, L=2048, E=1024, H=16, HW=64.
// Roofline: ~39 GFLOP vs 256MB bias stream -> HBM-bound. bf16 WMMA + f32 acc;
// bias double-buffered through LDS with GLOBAL_LOAD_ASYNC_TO_LDS (every 64B
// line fetched once and shared by all 16 heads in the workgroup); GEMM A-tiles
// double-buffered via the Tensor Data Mover (pad_enable reproduces the padded
// LDS stride), so both DMA streams run under the WMMA compute.
// ---------------------------------------------------------------------------

#define L_SEQ 2048
#define EMB   1024
#define NH    16
#define HWID  64
#define E3    3072
#define QSCALE 0.125f   // HW^-0.5 = 1/8

typedef __attribute__((ext_vector_type(16))) __bf16 bf16x16;
typedef __attribute__((ext_vector_type(8)))  __bf16 bf16x8;
typedef __attribute__((ext_vector_type(8)))  float  f32x8;
typedef __attribute__((ext_vector_type(4)))  unsigned int u32x4;
typedef __attribute__((ext_vector_type(8)))  unsigned int u32x8;

__device__ inline bf16x16 join8(bf16x8 lo, bf16x8 hi) {
  union { bf16x16 v; bf16x8 h[2]; } u;
  u.h[0] = lo; u.h[1] = hi;
  return u.v;
}

// --- gfx1250 async/TDM helpers (inline asm; builtins not portable) ---------

// Per-lane async copy of 16 bytes global -> LDS (tracked by ASYNCcnt).
// LDS operand = low 32 bits of the flat address (LDS aperture maps addr[31:0]).
__device__ inline void async_ld_b128(void* lds_dst, const void* gsrc) {
  asm volatile("global_load_async_to_lds_b128 %0, %1, off"
               :: "v"((unsigned)(uintptr_t)lds_dst),
                  "v"((unsigned long long)(uintptr_t)gsrc)
               : "memory");
}
__device__ inline void wait_async0() {
  asm volatile("s_wait_asynccnt 0x0" ::: "memory");
}
// ASYNCcnt decrements in issue order: <=4 outstanding means the previous
// buffer's 4 chunks have fully retired while the next 4 stay in flight.
__device__ inline void wait_async4() {
  asm volatile("s_wait_asynccnt 0x4" ::: "memory");
}

// TDM 2-D tile load: tile_d1 rows x tile_d0 bf16, memory row stride stride_d0
// elems; pad_enable inserts 16B after every 64B stored -> LDS row stride 80B
// = 40 bf16 (matches the WMMA fragment-read layout). Descriptor per ISA 8.3/8.4.
__device__ inline void tdm_load_tile_to_lds(const void* gsrc, void* lds_dst,
                                            unsigned tensor_d0, unsigned tensor_d1,
                                            unsigned tile_d0, unsigned tile_d1,
                                            unsigned stride_d0) {
  unsigned long long ga = (unsigned long long)(uintptr_t)gsrc;
  u32x4 g0;
  g0[0] = 1u;                                        // count=1 (valid), user mode
  g0[1] = (unsigned)(uintptr_t)lds_dst;              // lds_addr
  g0[2] = (unsigned)(ga & 0xffffffffu);              // global_addr[31:0]
  g0[3] = (unsigned)((ga >> 32) & 0x01ffffffu)       // global_addr[56:32]
        | (2u << 30);                                // type = 2 ("image")
  u32x8 g1;
  g1[0] = (1u << 16)                                 // data_size = 2 bytes
        | (1u << 20)                                 // pad_enable
        | (3u << 22)                                 // pad_interval: 16 dwords
        | (3u << 25);                                // pad_amount: 4 dwords
  g1[1] = (tensor_d0 & 0xffffu) << 16;               // tensor_dim0 @ bit48
  g1[2] = (tensor_d0 >> 16) | ((tensor_d1 & 0xffffu) << 16);
  g1[3] = (tensor_d1 >> 16) | (tile_d0 << 16);       // tile_dim0 @ bit112
  g1[4] = tile_d1;                                   // tile_dim1 (tile_dim2 = 0)
  g1[5] = stride_d0;                                 // tensor_dim0_stride[31:0]
  g1[6] = 0u;
  g1[7] = 0u;
  asm volatile("tensor_load_to_lds %0, %1" :: "s"(g0), "s"(g1) : "memory");
}

// ---------------------------------------------------------------------------
// Elementwise converters
// ---------------------------------------------------------------------------
__global__ void k_f32_to_bf16(const float* __restrict__ s, __bf16* __restrict__ d, int n) {
  int i = blockIdx.x * blockDim.x + threadIdx.x;
  if (i < n) d[i] = (__bf16)s[i];
}

__global__ void k_transpose_bf16(const float* __restrict__ w, __bf16* __restrict__ wt,
                                 int N, int K) {
  int i = blockIdx.x * blockDim.x + threadIdx.x;
  if (i < N * K) {
    int n = i / K, k = i % K;
    wt[(size_t)k * N + n] = (__bf16)w[i];
  }
}

// ---------------------------------------------------------------------------
// Tiled GEMM: C[M,N] = A[M,K](bf16) * B[K,N](bf16) (+ bias[n]), fp32 out.
// 256 threads = 8 waves; block tile 128x16, K stepped by 32. A tiles are
// double-buffered via TDM (DMA of tile kb+32 overlaps WMMA on tile kb);
// B tile transpose-scattered by threads. Fragment layouts per ISA 7.12.2.
// ---------------------------------------------------------------------------
__global__ __launch_bounds__(256)
void k_gemm_bf16(const __bf16* __restrict__ A, const __bf16* __restrict__ B,
                 const float* __restrict__ bias, float* __restrict__ C,
                 int M, int N, int K) {
  __shared__ __attribute__((aligned(16))) __bf16 As[2][128 * 40];
  __shared__ __attribute__((aligned(16))) __bf16 Bt[16 * 40];

  const int tid  = threadIdx.x;
  const int wave = tid >> 5;
  const int lg   = (tid >> 4) & 1;
  const int lm   = tid & 15;
  const int m0   = blockIdx.y * 128;
  const int n0   = blockIdx.x * 16;

  // Prologue: DMA the first A tile.
  if (wave == 0) {
    tdm_load_tile_to_lds(A + (size_t)m0 * K, As[0],
                         (unsigned)K, (unsigned)(M - m0), 32u, 128u, (unsigned)K);
  }

  f32x8 acc = {};
  for (int kb = 0; kb < K; kb += 32) {
    const int buf = (kb >> 5) & 1;
    // Issue DMA for the next A tile into the other buffer, then wait for the
    // current one (tensor ops are in-order per wave -> tensorcnt<=1 suffices).
    if (wave == 0) {
      if (kb + 32 < K) {
        tdm_load_tile_to_lds(A + (size_t)m0 * K + kb + 32, As[buf ^ 1],
                             (unsigned)K, (unsigned)(M - m0), 32u, 128u, (unsigned)K);
        __builtin_amdgcn_s_wait_tensorcnt(1);
      } else {
        __builtin_amdgcn_s_wait_tensorcnt(0);
      }
    }
    // B tile transposed into LDS: [k][n] -> Bt[n][k].
    {
      int idx = tid * 2;
      int k = idx >> 4, n = idx & 15;
      const __bf16* src = B + (size_t)(kb + k) * N + n0 + n;
      Bt[n * 40 + k]       = src[0];
      Bt[(n + 1) * 40 + k] = src[1];
    }
    __syncthreads();

    int mloc = wave * 16 + lm;
    bf16x8 alo = *(const bf16x8*)&As[buf][mloc * 40 + lg * 8];
    bf16x8 ahi = *(const bf16x8*)&As[buf][mloc * 40 + 16 + lg * 8];
    bf16x16 af = join8(alo, ahi);
    bf16x8 blo = *(const bf16x8*)&Bt[lm * 40 + lg * 8];
    bf16x8 bhi = *(const bf16x8*)&Bt[lm * 40 + 16 + lg * 8];
    bf16x16 bfr = join8(blo, bhi);

    acc = __builtin_amdgcn_wmma_f32_16x16x32_bf16(false, af, false, bfr,
                                                  (short)0, acc, false, false);
    __syncthreads();   // all reads of As[buf]/Bt done before next overwrite
  }

  int n = n0 + lm;
  float badd = bias ? bias[n] : 0.f;
#pragma unroll
  for (int r = 0; r < 8; ++r) {
    int m = m0 + wave * 16 + r + 8 * lg;
    C[(size_t)m * N + n] = acc[r] + badd;
  }
}

// ---------------------------------------------------------------------------
// Repack QKV gemm output [L][H][3*HW] into per-head bf16 tensors:
// Q [H][L][HW] (pre-scaled), K [H][L][HW], V^T [H][HW][L].
// ---------------------------------------------------------------------------
__global__ void k_qkv_post(const float* __restrict__ t, __bf16* __restrict__ qb,
                           __bf16* __restrict__ km, __bf16* __restrict__ vt) {
  int i = blockIdx.x * blockDim.x + threadIdx.x;   // over L*E
  int l = i >> 10;
  int rem = i & 1023;
  int h = rem >> 6;
  int c = rem & 63;
  const float* base = t + (size_t)l * E3 + h * 192;
  qb[((size_t)h * L_SEQ + l) * HWID + c] = (__bf16)(QSCALE * base[c]);
  km[((size_t)h * L_SEQ + l) * HWID + c] = (__bf16)(base[64 + c]);
  vt[((size_t)h * HWID + c) * L_SEQ + l] = (__bf16)(base[128 + c]);
}

// ---------------------------------------------------------------------------
// Flash attention: 1 workgroup = 16 waves = all 16 heads of one 16-row q tile.
// The bias tile [16 q][32 k][16 h] (32KB) is double-buffered in LDS via async
// b128 loads: each iteration issues the NEXT block's chunks, then waits
// asynccnt<=4 so the current buffer is ready while the next streams in.
// ---------------------------------------------------------------------------
__device__ inline void stage_bias_tile(float* dst, const float* __restrict__ bias,
                                       int q0, int kb) {
#pragma unroll
  for (int j = 0; j < 4; ++j) {
    int c   = (int)threadIdx.x + 512 * j;    // 16B chunk id
    int m   = c >> 7;                        // 128 chunks per q-row
    int off = (c & 127) << 2;                // float offset within row
    async_ld_b128(&dst[m * 512 + off],
                  bias + ((size_t)(q0 + m) * L_SEQ + kb) * NH + off);
  }
}

__global__ __launch_bounds__(512)
void k_attn(const __bf16* __restrict__ Q, const __bf16* __restrict__ Km,
            const __bf16* __restrict__ Vt, const float* __restrict__ bias,
            const unsigned char* __restrict__ mask, float* __restrict__ Y) {
  __shared__ __attribute__((aligned(16))) float  bias_s[2][16 * 512]; // [m][key][h]
  __shared__ __attribute__((aligned(16))) __bf16 pls[16 * 640];       // per-wave P
  const int h    = threadIdx.x >> 5;
  const int lane = threadIdx.x & 31;
  const int lg   = lane >> 4;
  const int lm   = lane & 15;
  const int q0   = blockIdx.x * 16;
  __bf16* P = &pls[h * 640];

  // Q fragments (A layout), channel steps [0,32) and [32,64). Kept in regs.
  bf16x16 qa[2];
  {
    const __bf16* qrow = Q + ((size_t)h * L_SEQ + q0 + lm) * HWID;
#pragma unroll
    for (int s = 0; s < 2; ++s) {
      bf16x8 lo = *(const bf16x8*)(qrow + s * 32 + lg * 8);
      bf16x8 hi = *(const bf16x8*)(qrow + s * 32 + 16 + lg * 8);
      qa[s] = join8(lo, hi);
    }
  }

  f32x8 acc[4] = {{}, {}, {}, {}};
  float mrow[8], lrow[8];
#pragma unroll
  for (int r = 0; r < 8; ++r) { mrow[r] = -1e30f; lrow[r] = 0.f; }

  // Prologue: stream the first bias block.
  stage_bias_tile(bias_s[0], bias, q0, 0);

  for (int kb = 0; kb < L_SEQ; kb += 32) {
    const int buf = (kb >> 5) & 1;
    if (kb + 32 < L_SEQ) {
      __builtin_prefetch(bias + ((size_t)q0 * L_SEQ + kb + 32) * NH, 0, 0);
      stage_bias_tile(bias_s[buf ^ 1], bias, q0, kb + 32);
      wait_async4();                 // current buffer retired, next in flight
    } else {
      wait_async0();
    }
    __syncthreads();

    f32x8 s01[2];
#pragma unroll
    for (int t = 0; t < 2; ++t) {
      int key = kb + t * 16 + lm;
      const __bf16* krow = Km + ((size_t)h * L_SEQ + key) * HWID;
      f32x8 c = {};
#pragma unroll
      for (int s = 0; s < 2; ++s) {
        bf16x8 lo = *(const bf16x8*)(krow + s * 32 + lg * 8);
        bf16x8 hi = *(const bf16x8*)(krow + s * 32 + 16 + lg * 8);
        bf16x16 kf = join8(lo, hi);
        c = __builtin_amdgcn_wmma_f32_16x16x32_bf16(false, qa[s], false, kf,
                                                    (short)0, c, false, false);
      }
      int mk = mask[key];
      int kl = t * 16 + lm;                    // key local to block
#pragma unroll
      for (int r = 0; r < 8; ++r) {
        int m = r + 8 * lg;
        float v = c[r] + bias_s[buf][m * 512 + kl * 16 + h];
        c[r] = mk ? v : -1e30f;
      }
      s01[t] = c;
    }
    __syncthreads();   // bias_s[buf] fully consumed; next iter may overwrite it

    // Online softmax across the 32-key block (row reduce = width-16 butterfly)
#pragma unroll
    for (int r = 0; r < 8; ++r) {
      float tmax = fmaxf(s01[0][r], s01[1][r]);
#pragma unroll
      for (int d = 1; d < 16; d <<= 1) tmax = fmaxf(tmax, __shfl_xor(tmax, d, 16));
      float nm = fmaxf(mrow[r], tmax);
      float scale = __expf(mrow[r] - nm);
      float p0 = __expf(s01[0][r] - nm);
      float p1 = __expf(s01[1][r] - nm);
      float rs = p0 + p1;
#pragma unroll
      for (int d = 1; d < 16; d <<= 1) rs += __shfl_xor(rs, d, 16);
      lrow[r] = lrow[r] * scale + rs;
      mrow[r] = nm;
      int m = r + 8 * lg;
      P[m * 40 + lm]      = (__bf16)p0;        // C layout -> LDS [row][key]
      P[m * 40 + 16 + lm] = (__bf16)p1;
#pragma unroll
      for (int nt = 0; nt < 4; ++nt) acc[nt][r] *= scale;
    }

    // P as A fragment (16 rows x 32 keys); wave-private LDS, in-order per wave
    bf16x8 plo = *(const bf16x8*)&P[lm * 40 + lg * 8];
    bf16x8 phi = *(const bf16x8*)&P[lm * 40 + 16 + lg * 8];
    bf16x16 pf = join8(plo, phi);

#pragma unroll
    for (int nt = 0; nt < 4; ++nt) {
      const __bf16* vrow = Vt + ((size_t)h * HWID + nt * 16 + lm) * L_SEQ + kb;
      bf16x8 vlo = *(const bf16x8*)(vrow + lg * 8);
      bf16x8 vhi = *(const bf16x8*)(vrow + 16 + lg * 8);
      bf16x16 vf = join8(vlo, vhi);
      acc[nt] = __builtin_amdgcn_wmma_f32_16x16x32_bf16(false, pf, false, vf,
                                                        (short)0, acc[nt], false, false);
    }
  }

  // Normalize and write y[q][h*64 + c] (fp32)
#pragma unroll
  for (int r = 0; r < 8; ++r) {
    float inv = 1.f / lrow[r];
    int m = q0 + r + 8 * lg;
#pragma unroll
    for (int nt = 0; nt < 4; ++nt)
      Y[(size_t)m * EMB + h * HWID + nt * 16 + lm] = acc[nt][r] * inv;
  }
}

// ---------------------------------------------------------------------------
__global__ void k_gate(const float* __restrict__ gt, const float* __restrict__ Yin,
                       __bf16* __restrict__ yb, int n) {
  int i = blockIdx.x * blockDim.x + threadIdx.x;
  if (i < n) {
    float g = 1.f / (1.f + __expf(-gt[i]));
    yb[i] = (__bf16)(g * Yin[i]);
  }
}

// ---------------------------------------------------------------------------
extern "C" void kernel_launch(void* const* d_in, const int* in_sizes, int n_in,
                              void* d_out, int out_size, void* d_ws, size_t ws_size,
                              hipStream_t stream) {
  const float*         x      = (const float*)d_in[0];
  const unsigned char* mask   = (const unsigned char*)d_in[1];  // jax bool = u8
  const float*         bias   = (const float*)d_in[2];
  const float*         w_proj = (const float*)d_in[3];
  const float*         w_o    = (const float*)d_in[4];
  const float*         b_o    = (const float*)d_in[5];
  const float*         w_g    = (const float*)d_in[6];
  const float*         b_g    = (const float*)d_in[7];
  float* out = (float*)d_out;

  char* p = (char*)d_ws;
  auto alloc = [&](size_t bytes) -> char* {
    char* r = p;
    p += (bytes + 255) & ~(size_t)255;
    return r;
  };
  __bf16* xb  = (__bf16*)alloc((size_t)L_SEQ * EMB * 2);
  __bf16* wpT = (__bf16*)alloc((size_t)EMB * E3 * 2);
  __bf16* wgT = (__bf16*)alloc((size_t)EMB * EMB * 2);
  __bf16* woT = (__bf16*)alloc((size_t)EMB * EMB * 2);
  __bf16* qb  = (__bf16*)alloc((size_t)NH * L_SEQ * HWID * 2);
  __bf16* km  = (__bf16*)alloc((size_t)NH * L_SEQ * HWID * 2);
  __bf16* vt  = (__bf16*)alloc((size_t)NH * HWID * L_SEQ * 2);
  char*   tmp = alloc((size_t)L_SEQ * E3 * 4);           // 24MB, time-shared
  float*  t_qkv = (float*)tmp;                           //  phase 1: [L,3E]
  float*  gt    = (float*)tmp;                           //  phase 2: gate [L,E]
  float*  Yat   = (float*)(tmp + (size_t)L_SEQ * EMB * 4);
  __bf16* yb    = (__bf16*)(tmp + (size_t)2 * L_SEQ * EMB * 4);

  const int TPB = 256;
  k_f32_to_bf16<<<(L_SEQ * EMB) / TPB, TPB, 0, stream>>>(x, xb, L_SEQ * EMB);
  k_transpose_bf16<<<(E3 * EMB) / TPB, TPB, 0, stream>>>(w_proj, wpT, E3, EMB);
  k_transpose_bf16<<<(EMB * EMB) / TPB, TPB, 0, stream>>>(w_g, wgT, EMB, EMB);
  k_transpose_bf16<<<(EMB * EMB) / TPB, TPB, 0, stream>>>(w_o, woT, EMB, EMB);

  // t = x @ w_proj^T  [L, 3E]
  k_gemm_bf16<<<dim3(E3 / 16, L_SEQ / 128), 256, 0, stream>>>(
      xb, wpT, nullptr, t_qkv, L_SEQ, E3, EMB);
  k_qkv_post<<<(L_SEQ * EMB) / TPB, TPB, 0, stream>>>(t_qkv, qb, km, vt);

  // gate = x @ w_g^T + b_g  [L, E]
  k_gemm_bf16<<<dim3(EMB / 16, L_SEQ / 128), 256, 0, stream>>>(
      xb, wgT, b_g, gt, L_SEQ, EMB, EMB);

  // fused bias + mask + softmax + PV
  k_attn<<<L_SEQ / 16, 512, 0, stream>>>(qb, km, vt, bias, mask, Yat);

  // y = sigmoid(gate) * y
  k_gate<<<(L_SEQ * EMB) / TPB, TPB, 0, stream>>>(gt, Yat, yb, L_SEQ * EMB);

  // out = y @ w_o^T + b_o
  k_gemm_bf16<<<dim3(EMB / 16, L_SEQ / 128), 256, 0, stream>>>(
      yb, woT, b_o, out, L_SEQ, EMB, EMB);
}
